// CombinedConsecutiveAdjustment_42580305772649
// MI455X (gfx1250) — compile-verified, matrix operand order checked by reading, not run on hardware
//
#include <hip/hip_runtime.h>
#include <math.h>

// D = A(16x64 u8) x B(64x16 u8) + C(16x16 i32), one matrix per wave32.
typedef __attribute__((ext_vector_type(8))) int v8i;

#define SEQ    4096
#define WAVES  4
#define NTHR   (WAVES * 32)     // 128 threads / block, 1 block per row

__global__ __launch_bounds__(NTHR)
void CombinedConsecutiveAdjustment_kernel(const float* __restrict__ drowsy,
                                          const int*   __restrict__ gest,
                                          float*       __restrict__ out)
{
    const int row  = blockIdx.x;
    const int lane = threadIdx.x & 31;
    const int wave = threadIdx.x >> 5;

    // Row base; each wave covers 1024 consecutive elements, lane-coalesced
    // int4 loads: chunk c, lane l -> int4 index (wave*256 + c*32 + l).
    const int4* gp = reinterpret_cast<const int4*>(gest) + (size_t)row * (SEQ / 4);
    const int   baseVi = wave * 256 + lane;          // int4 index of chunk 0

    // --- Load + pack: 1024 u8 values per wave = the WMMA A operand -------------
    // g values are guaranteed 0/1, so no masking needed; v_perm_b32 packs the
    // low byte of each of the 4 dwords into one byte-packed dword (3 VALU/chunk).
    v8i a;
#pragma unroll
    for (int c = 0; c < 8; ++c) {
        const int4 v = gp[baseVi + c * 32];          // global_load_b128, coalesced
        const unsigned lo = __byte_perm((unsigned)v.x, (unsigned)v.y, 0x0C0C0400u); // bytes {0,0,y0,x0}
        const unsigned hi = __byte_perm((unsigned)v.z, (unsigned)v.w, 0x04000C0Cu); // bytes {w0,z0,0,0}
        a[c] = (int)(lo | hi);                       // byte j = g[e+j], e = (baseVi+c*32)*4
    }

    // --- Count of ones via one matrix op --------------------------------------
    // B = all-ones 64x16 u8 -> D[m][n] = sum_k A[m][k]; summing one full column
    // of D gives the total over all 1024 bytes. EXEC is all-ones here.
    v8i b;
#pragma unroll
    for (int i = 0; i < 8; ++i) b[i] = 0x01010101;
    v8i c_acc = {};
    c_acc = __builtin_amdgcn_wmma_i32_16x16x64_iu8(false, a, false, b, c_acc, false, false);

    // --- Last-one index from the packed words (independent of WMMA D regs; this
    // chain fills the IU8 WMMA->VALU hazard slots) ------------------------------
    // Bytes are 0/1, so highest set bit of pk is at bit 8*j -> j = 3 - (clz>>3).
    // Chunks are ascending in element index per lane, so overwrite-if-nonzero.
    int lastIdx = -1;
#pragma unroll
    for (int c = 0; c < 8; ++c) {
        const unsigned pk = (unsigned)a[c];
        const int e3 = (baseVi + c * 32) * 4 + 3;    // index of byte 3 in this chunk
        const int cand = e3 - (__clz(pk) >> 3);      // valid only when pk != 0
        if (pk) lastIdx = cand;
    }
#pragma unroll
    for (int off = 16; off >= 1; off >>= 1)
        lastIdx = max(lastIdx, __shfl_xor(lastIdx, off, 32));

    // --- Fold the D matrix: per-lane half-column sum, then pair lanes n, n^16 --
    int s = c_acc[0] + c_acc[1] + c_acc[2] + c_acc[3] +
            c_acc[4] + c_acc[5] + c_acc[6] + c_acc[7];
    const int waveCount = s + __shfl_xor(s, 16, 32); // total ones in wave's 1024 elems

    __shared__ int s_cnt[WAVES];
    __shared__ int s_li[WAVES];
    if (lane == 0) { s_cnt[wave] = waveCount; s_li[wave] = lastIdx; }
    __syncthreads();

    if (threadIdx.x == 0) {
        int nOnes = 0, pos = -1;
#pragma unroll
        for (int w = 0; w < WAVES; ++w) { nOnes += s_cnt[w]; pos = max(pos, s_li[w]); }

        // max_eye = nOnes; pos = last one-index; max_att = (S-1)-pos
        // (pos==S-1 -> 0, fails the >=40 gate, matching the -inf case).
        float adjustment = 0.0f;
        if (nOnes >= 40 && pos >= 0) {
            const int maxAtt = (SEQ - 1) - pos;
            if (maxAtt >= 40) {
                const float excess = (float)(maxAtt - 40);
                adjustment = 0.05f * (1.0f - expf(-excess / (160.0f / 3.0f)));
            }
        }
        float o = drowsy[row] * (1.0f - adjustment);
        o = fminf(fmaxf(o, 0.01f), 1.0f);
        out[row] = o;
    }
}

extern "C" void kernel_launch(void* const* d_in, const int* in_sizes, int n_in,
                              void* d_out, int out_size, void* d_ws, size_t ws_size,
                              hipStream_t stream)
{
    const float* drowsy = (const float*)d_in[0];   // [B, 1] float32
    const int*   gest   = (const int*)d_in[1];     // [B, S, 1] int32 (0/1)
    float*       out    = (float*)d_out;           // [B, 1] float32

    const int B = in_sizes[0];                     // 8192 rows; S fixed at 4096
    CombinedConsecutiveAdjustment_kernel<<<B, NTHR, 0, stream>>>(drowsy, gest, out);
}
